// BENDRContext_21088289423852
// MI455X (gfx1250) — matrix operand order repaired and checked by reading, not compiled, not run on hardware
//
#include <hip/hip_runtime.h>

#define BB 4
#define EE 256
#define SS 2048
#define NH 8
#define DH 32
#define FF 1024
#define NL 4
#define ROWS (BB * SS) /* 8192 */

typedef _Float16 half8  __attribute__((ext_vector_type(8)));
typedef _Float16 half16 __attribute__((ext_vector_type(16)));
typedef float    float8 __attribute__((ext_vector_type(8)));

__device__ inline float8 f8zero() {
  float8 z;
#pragma unroll
  for (int i = 0; i < 8; ++i) z[i] = 0.f;
  return z;
}

__device__ inline half16 cat8(half8 lo, half8 hi) {
  half16 r;
#pragma unroll
  for (int i = 0; i < 8; ++i) { r[i] = lo[i]; r[i + 8] = hi[i]; }
  return r;
}

// A-matrix fragment (16x32 f16). Lane l holds row m0+(l&15);
// elements 0..7 = K k0+base..base+7, elements 8..15 = K k0+16+base..+23, base = (l<16)?0:8.
__device__ inline half16 load_a_frag(const _Float16* A, int stride, int m0, int k0, int lane) {
  int m = m0 + (lane & 15);
  int base = (lane < 16) ? 0 : 8;
  const _Float16* p = A + (size_t)m * stride + k0 + base;
  half8 lo = *(const half8*)(p);
  half8 hi = *(const half8*)(p + 16);
  return cat8(lo, hi);
}

// B-matrix fragment (32x16 f16) from Bt stored row-major [N][K]:
// lane l holds column n0+(l&15); elements 0..7 = K k0+base..base+7, 8..15 = K k0+16+base..+23, base = (l<16)?0:8.
__device__ inline half16 load_bt_frag(const _Float16* Bt, int stride, int n0, int k0, int lane) {
  int n = n0 + (lane & 15);
  int koff = k0 + ((lane >> 4) << 3);
  const _Float16* p = Bt + (size_t)n * stride + koff;
  half8 lo = *(const half8*)(p);
  half8 hi = *(const half8*)(p + 16);
  return cat8(lo, hi);
}

__device__ inline float8 wmma16(half16 a, half16 b, float8 c) {
  return __builtin_amdgcn_wmma_f32_16x16x32_f16(false, a, false, b, (short)0, c, false, false);
}

__device__ inline float hred_max(float v) {
#pragma unroll
  for (int o = 8; o; o >>= 1) v = fmaxf(v, __shfl_xor(v, o, 32));
  return v;
}
__device__ inline float hred_sum(float v) {
#pragma unroll
  for (int o = 8; o; o >>= 1) v += __shfl_xor(v, o, 32);
  return v;
}
__device__ inline float wred_sum(float v) {
#pragma unroll
  for (int o = 16; o; o >>= 1) v += __shfl_xor(v, o, 32);
  return v;
}

// ---------------- weight conversion ----------------
__global__ void k_cvt_f16(const float* __restrict__ src, _Float16* __restrict__ dst, int n) {
  int i = blockIdx.x * 256 + threadIdx.x;
  if (i < n) dst[i] = (_Float16)src[i];
}

// Packed transposed QKV weights: dst[l][n][k], n = sel*256 + head*32 + d (q scaled).
__global__ void k_cvt_qkv(const float* __restrict__ wq, const float* __restrict__ wk,
                          const float* __restrict__ wv, _Float16* __restrict__ dst, float qs) {
  int idx = blockIdx.x * 256 + threadIdx.x;
  if (idx >= NL * 768 * 256) return;
  int k = idx & 255;
  int n = (idx >> 8) % 768;
  int l = (idx >> 8) / 768;
  int sel = n >> 8;
  int nc = n & 255;
  int head = nc >> 5;
  int d = nc & 31;
  const float* w = (sel == 0) ? wq : ((sel == 1) ? wk : wv);
  float v = w[(((size_t)(l * NH + head)) * EE + k) * DH + d];
  if (sel == 0) v *= qs;
  dst[idx] = (_Float16)v;
}

// ---------------- conv encoder ----------------
__global__ void k_conv(const float* __restrict__ x, const float* __restrict__ cw,
                       const float* __restrict__ cb, float* __restrict__ hc) {
  int idx = blockIdx.x * 256 + threadIdx.x; // [b][e][s]
  int s = idx & 2047;
  int e = (idx >> 11) & 255;
  int b = idx >> 19;
  float acc = cb[e];
  int t = s * 2 - 1;
#pragma unroll 4
  for (int ci = 0; ci < 20; ++ci) {
    const float* xr = x + ((size_t)(b * 20 + ci)) * 4096;
    const float* wr = cw + (e * 20 + ci) * 3;
    if (t >= 0) acc += xr[t] * wr[0];
    acc += xr[t + 1] * wr[1];
    acc += xr[t + 2] * wr[2];
  }
  hc[idx] = acc;
}

__global__ void k_gn_stats(const float* __restrict__ hc, float* __restrict__ gmean,
                           float* __restrict__ grstd) {
  __shared__ float ssum[256], ssq[256];
  int grp = blockIdx.x;           // b*128 + g
  int b = grp >> 7, g = grp & 127;
  float sum = 0.f, sq = 0.f;
  for (int i = threadIdx.x; i < 4096; i += 256) {
    int c = i >> 11, s = i & 2047;
    float v = hc[((size_t)(b * EE + g * 2 + c)) * SS + s];
    sum += v; sq += v * v;
  }
  ssum[threadIdx.x] = sum; ssq[threadIdx.x] = sq;
  __syncthreads();
  for (int st = 128; st; st >>= 1) {
    if (threadIdx.x < st) { ssum[threadIdx.x] += ssum[threadIdx.x + st]; ssq[threadIdx.x] += ssq[threadIdx.x + st]; }
    __syncthreads();
  }
  if (threadIdx.x == 0) {
    float m = ssum[0] * (1.f / 4096.f);
    float var = ssq[0] * (1.f / 4096.f) - m * m;
    gmean[grp] = m;
    grstd[grp] = rsqrtf(var + 1e-5f);
  }
}

// GN apply + GELU + transpose + mask replace + pos encoding -> h fp32 [b][s][e] + f16 copy
__global__ void k_fuse(const float* __restrict__ hc, const float* __restrict__ gmean,
                       const float* __restrict__ grstd, const float* __restrict__ gg,
                       const float* __restrict__ gb, const unsigned char* __restrict__ mask,
                       const float* __restrict__ mrepl, float* __restrict__ hout,
                       _Float16* __restrict__ hbout) {
  int idx = blockIdx.x * 256 + threadIdx.x; // [b][s][e]
  int e = idx & 255;
  int s = (idx >> 8) & 2047;
  int b = idx >> 19;
  float v = hc[((size_t)(b * EE + e)) * SS + s];
  int grp = b * 128 + (e >> 1);
  v = (v - gmean[grp]) * grstd[grp];
  v = v * gg[e] + gb[e];
  v = 0.5f * v * (1.f + erff(v * 0.70710678118f));
  if (mask[b * SS + s]) v = mrepl[e];
  float ang = (float)s * powf(10000.f, -(float)e * (1.f / 256.f));
  v += ((e & 1) == 0) ? sinf(ang) : cosf(ang);
  hout[idx] = v;
  hbout[idx] = (_Float16)v;
}

// ---------------- generic WMMA GEMM: C[M,N] = A[M,K] * Bt[N,K]^T (+bias)(+residual) -------
__global__ __launch_bounds__(256) void k_gemm(const _Float16* __restrict__ A,
                                              const _Float16* __restrict__ Bt,
                                              int M, int N, int K,
                                              const float* __restrict__ bias,
                                              const float* __restrict__ residual,
                                              float* __restrict__ outF,
                                              _Float16* __restrict__ outH) {
  int lane = threadIdx.x & 31;
  int wid = blockIdx.x * 8 + (threadIdx.x >> 5);
  int ntn = N >> 6;
  int total = (M >> 4) * ntn;
  if (wid >= total) return;                // wave-uniform
  int mt = wid / ntn, nt = wid % ntn;
  int m0 = mt << 4, n0 = nt << 6;
  float8 acc[4];
#pragma unroll
  for (int t = 0; t < 4; ++t) acc[t] = f8zero();
  for (int k0 = 0; k0 < K; k0 += 32) {
    half16 a = load_a_frag(A, K, m0, k0, lane);
#pragma unroll
    for (int t = 0; t < 4; ++t) {
      half16 b = load_bt_frag(Bt, K, n0 + t * 16, k0, lane);
      acc[t] = wmma16(a, b, acc[t]);
    }
  }
  int hof = (lane >> 4) << 3;
  int nn = lane & 15;
#pragma unroll
  for (int j = 0; j < 8; ++j) {
    int m = m0 + j + hof;
#pragma unroll
    for (int t = 0; t < 4; ++t) {
      int col = n0 + t * 16 + nn;
      float v = acc[t][j];
      if (bias) v += bias[col];
      size_t idx = (size_t)m * N + col;
      if (residual) v += residual[idx];
      if (outF) outF[idx] = v;
      if (outH) outH[idx] = (_Float16)v;
    }
  }
}

// ---------------- flash attention ----------------
__global__ __launch_bounds__(128) void k_attn(const _Float16* __restrict__ qkv,
                                              _Float16* __restrict__ aout) {
  __shared__ __align__(16) _Float16 Kt[64 * 32]; // [key][d]
  __shared__ __align__(16) _Float16 Vt[32 * 64]; // [d][key] (transposed)
  __shared__ __align__(16) _Float16 Pl[4 * 16 * 32];
  int tid = threadIdx.x, lane = tid & 31, wv = tid >> 5;
  int bid = blockIdx.x;
  int qt = bid & 31, hh = (bid >> 5) & 7, b = bid >> 8;
  int qrow0 = b * SS + qt * 64 + wv * 16;

  half16 aq = load_a_frag(qkv + hh * 32, 768, qrow0, 0, lane);

  float mprev[8], lsum[8];
  float8 o0 = f8zero(), o1 = f8zero();
#pragma unroll
  for (int j = 0; j < 8; ++j) { mprev[j] = -1e30f; lsum[j] = 0.f; }

  int hof = (lane >> 4) << 3;
  int nn = lane & 15;
  _Float16* pw = Pl + wv * 512;

  for (int kv0 = 0; kv0 < SS; kv0 += 64) {
    __syncthreads();
#pragma unroll
    for (int it = 0; it < 2; ++it) {
      int c = tid * 2 + it;                 // 0..255
      int kv = c >> 2, d0 = (c & 3) << 3;
      size_t grow = (size_t)(b * SS + kv0 + kv) * 768 + hh * 32 + d0;
      half8 kk = *(const half8*)(qkv + grow + 256);
      *(half8*)(&Kt[kv * 32 + d0]) = kk;
      half8 vvv = *(const half8*)(qkv + grow + 512);
#pragma unroll
      for (int i = 0; i < 8; ++i) Vt[(d0 + i) * 64 + kv] = vvv[i];
    }
    __syncthreads();

#pragma unroll
    for (int sub = 0; sub < 2; ++sub) {
      half16 bk0 = load_bt_frag(Kt, 32, sub * 32, 0, lane);
      half16 bk1 = load_bt_frag(Kt, 32, sub * 32 + 16, 0, lane);
      float8 s0 = wmma16(aq, bk0, f8zero());
      float8 s1 = wmma16(aq, bk1, f8zero());
#pragma unroll
      for (int j = 0; j < 8; ++j) {
        float tm = hred_max(fmaxf(s0[j], s1[j]));
        float mnew = fmaxf(mprev[j], tm);
        float alpha = expf(mprev[j] - mnew);
        float p0 = expf(s0[j] - mnew);
        float p1 = expf(s1[j] - mnew);
        float rs = hred_sum(p0 + p1);
        lsum[j] = lsum[j] * alpha + rs;
        mprev[j] = mnew;
        o0[j] *= alpha; o1[j] *= alpha;
        int m = j + hof;
        pw[m * 32 + nn] = (_Float16)p0;
        pw[m * 32 + 16 + nn] = (_Float16)p1;
      }
      half16 ap = load_a_frag(Pl + wv * 512, 32, 0, 0, lane);
      half16 bv0 = load_bt_frag(Vt, 64, 0, sub * 32, lane);
      half16 bv1 = load_bt_frag(Vt, 64, 16, sub * 32, lane);
      o0 = wmma16(ap, bv0, o0);
      o1 = wmma16(ap, bv1, o1);
    }
  }
#pragma unroll
  for (int j = 0; j < 8; ++j) {
    float inv = 1.f / lsum[j];
    size_t row = (size_t)(qrow0 + j + hof);
    aout[row * EE + hh * 32 + nn] = (_Float16)(o0[j] * inv);
    aout[row * EE + hh * 32 + 16 + nn] = (_Float16)(o1[j] * inv);
  }
}

// ---------------- layer norm (one wave per 256-wide row) ----------------
__global__ __launch_bounds__(256) void k_ln(const float* __restrict__ in,
                                            const float* __restrict__ g,
                                            const float* __restrict__ bta,
                                            float* __restrict__ outF,
                                            _Float16* __restrict__ outH, int rows) {
  int lane = threadIdx.x & 31;
  int row = blockIdx.x * 8 + (threadIdx.x >> 5);
  if (row >= rows) return;
  const float* x = in + (size_t)row * EE;
  float v[8];
  float sum = 0.f;
#pragma unroll
  for (int i = 0; i < 8; ++i) { v[i] = x[lane + i * 32]; sum += v[i]; }
  sum = wred_sum(sum);
  float mean = sum * (1.f / 256.f);
  float sq = 0.f;
#pragma unroll
  for (int i = 0; i < 8; ++i) { float d = v[i] - mean; sq += d * d; }
  sq = wred_sum(sq);
  float rstd = rsqrtf(sq * (1.f / 256.f) + 1e-5f);
#pragma unroll
  for (int i = 0; i < 8; ++i) {
    int e = lane + i * 32;
    float y = (v[i] - mean) * rstd * g[e] + bta[e];
    outF[(size_t)row * EE + e] = y;
    outH[(size_t)row * EE + e] = (_Float16)y;
  }
}

extern "C" void kernel_launch(void* const* d_in, const int* in_sizes, int n_in,
                              void* d_out, int out_size, void* d_ws, size_t ws_size,
                              hipStream_t stream) {
  const float* x        = (const float*)d_in[0];
  const unsigned char* mask = (const unsigned char*)d_in[1];
  const float* conv_w   = (const float*)d_in[2];
  const float* conv_b   = (const float*)d_in[3];
  const float* gn_g     = (const float*)d_in[4];
  const float* gn_b     = (const float*)d_in[5];
  const float* mrepl    = (const float*)d_in[6];
  const float* wq       = (const float*)d_in[7];
  const float* wk       = (const float*)d_in[8];
  const float* wv       = (const float*)d_in[9];
  const float* wo       = (const float*)d_in[10];
  const float* w1       = (const float*)d_in[11];
  const float* b1       = (const float*)d_in[12];
  const float* w2       = (const float*)d_in[13];
  const float* b2       = (const float*)d_in[14];
  const float* ln1_g    = (const float*)d_in[15];
  const float* ln1_b    = (const float*)d_in[16];
  const float* ln2_g    = (const float*)d_in[17];
  const float* ln2_b    = (const float*)d_in[18];
  float* out = (float*)d_out;

  char* ws = (char*)d_ws;
  size_t off = 0;
  auto wsa = [&](size_t bytes) -> void* {
    void* p = (void*)(ws + off);
    off += (bytes + 255) & ~(size_t)255;
    return p;
  };
  const size_t BES = (size_t)ROWS * EE;
  float*     h_conv = (float*)wsa(BES * 4);
  float*     hres   = (float*)wsa(BES * 4);
  float*     h1     = (float*)wsa(BES * 4);
  float*     tmp    = (float*)wsa(BES * 4);
  _Float16*  hb     = (_Float16*)wsa(BES * 2);
  _Float16*  h1b    = (_Float16*)wsa(BES * 2);
  _Float16*  abf    = (_Float16*)wsa(BES * 2);
  _Float16*  qkvb   = (_Float16*)wsa((size_t)ROWS * 768 * 2);
  _Float16*  fb     = (_Float16*)wsa((size_t)ROWS * FF * 2);
  float*     gmean  = (float*)wsa(BB * 128 * 4);
  float*     grstd  = (float*)wsa(BB * 128 * 4);
  _Float16*  wqkv_t = (_Float16*)wsa((size_t)NL * 768 * 256 * 2);
  _Float16*  wo_h   = (_Float16*)wsa((size_t)NL * EE * EE * 2);
  _Float16*  w1_h   = (_Float16*)wsa((size_t)NL * FF * EE * 2);
  _Float16*  w2_h   = (_Float16*)wsa((size_t)NL * EE * FF * 2);

  // weight conversion (q scaled by 1/sqrt(DH))
  k_cvt_qkv<<<(NL * 768 * 256) / 256, 256, 0, stream>>>(wq, wk, wv, wqkv_t, 0.1767766953f);
  k_cvt_f16<<<(NL * EE * EE) / 256, 256, 0, stream>>>(wo, wo_h, NL * EE * EE);
  k_cvt_f16<<<(NL * FF * EE) / 256, 256, 0, stream>>>(w1, w1_h, NL * FF * EE);
  k_cvt_f16<<<(NL * EE * FF) / 256, 256, 0, stream>>>(w2, w2_h, NL * EE * FF);

  // conv encoder
  k_conv<<<(int)(BES / 256), 256, 0, stream>>>(x, conv_w, conv_b, h_conv);
  k_gn_stats<<<BB * 128, 256, 0, stream>>>(h_conv, gmean, grstd);
  k_fuse<<<(int)(BES / 256), 256, 0, stream>>>(h_conv, gmean, grstd, gn_g, gn_b, mask,
                                               mrepl, hres, hb);

  for (int l = 0; l < NL; ++l) {
    // QKV projection: [8192,256] x [256,768]
    k_gemm<<<(512 * 12) / 8, 256, 0, stream>>>(hb, wqkv_t + (size_t)l * 768 * 256,
                                               ROWS, 768, 256, nullptr, nullptr, nullptr, qkvb);
    // flash attention
    k_attn<<<BB * NH * (SS / 64), 128, 0, stream>>>(qkvb, abf);
    // output projection + residual
    k_gemm<<<(512 * 4) / 8, 256, 0, stream>>>(abf, wo_h + (size_t)l * EE * EE,
                                              ROWS, EE, 256, nullptr, hres, tmp, nullptr);
    k_ln<<<ROWS / 8, 256, 0, stream>>>(tmp, ln1_g + l * EE, ln1_b + l * EE, h1, h1b, ROWS);
    // FFN (no nonlinearity between, per reference)
    k_gemm<<<(512 * 16) / 8, 256, 0, stream>>>(h1b, w1_h + (size_t)l * FF * EE,
                                               ROWS, FF, 256, b1 + l * FF, nullptr, nullptr, fb);
    k_gemm<<<(512 * 4) / 8, 256, 0, stream>>>(fb, w2_h + (size_t)l * EE * FF,
                                              ROWS, EE, FF, b2 + l * EE, h1, tmp, nullptr);
    float* hout = (l == NL - 1) ? out : hres;
    k_ln<<<ROWS / 8, 256, 0, stream>>>(tmp, ln2_g + l * EE, ln2_b + l * EE, hout, hb, ROWS);
  }
}
